// MultiHeadAttention_89335319757161
// MI455X (gfx1250) — matrix operand order repaired and checked by reading, not compiled, not run on hardware
//
#include <hip/hip_runtime.h>

// ---------------------------------------------------------------------------
// MI455X (gfx1250) fused multi-head attention, bf16 WMMA flash-attention style.
//   B=4, N=2048, D=512, H=8, HEAD=64, TEMP=8
// Round 3: packed bf16 converts, dword-granular A-tile builds, K/V prefetch.
// ---------------------------------------------------------------------------

typedef __attribute__((ext_vector_type(16))) __bf16 v16bf;
typedef __attribute__((ext_vector_type(2)))  __bf16 v2bf;
typedef __attribute__((ext_vector_type(8)))  float  v8f;
typedef __attribute__((ext_vector_type(2)))  float  v2f;
typedef __attribute__((ext_vector_type(8)))  unsigned v8u;

#define B_    4
#define N_    2048
#define D_    512
#define H_    8
#define HEAD_ 64
// softmax via exp2: fold log2(e)/TEMP into Q at conversion time
#define QSCALE 0.1803368801111244f   // log2(e) / 8.0

// ---- conversions ----------------------------------------------------------

// packed f32x2 -> bf16x2 (backend picks hw packed cvt when available)
__device__ __forceinline__ v2bf cvt2bf(float a, float b) {
  v2f t; t[0] = a; t[1] = b;
  return __builtin_convertvector(t, v2bf);
}
__device__ __forceinline__ unsigned cvt2bf_u32(float a, float b) {
  return __builtin_bit_cast(unsigned, cvt2bf(a, b));
}
// scalar fallback (staging transpose only)
__device__ __forceinline__ __bf16 f2bf(float x) {
  unsigned u = __builtin_bit_cast(unsigned, x);
  u += 0x7FFFu + ((u >> 16) & 1u);          // round-to-nearest-even
  unsigned short s = (unsigned short)(u >> 16);
  return __builtin_bit_cast(__bf16, s);
}

__device__ __forceinline__ v8f wmma_bf16(v16bf a, v16bf b, v8f c) {
  // 8 args: (neg_a, A, neg_b, B, c_mod, C, reuse_a, reuse_b)
  return __builtin_amdgcn_wmma_f32_16x16x32_bf16(false, a, false, b, (short)0, c,
                                                 false, false);
}

// ---- DPP16 cross-lane helpers (reduce across a 16-lane half, pure VALU) ----
template <int CTRL>
__device__ __forceinline__ float dpp_mov(float v) {
  return __builtin_bit_cast(
      float, __builtin_amdgcn_update_dpp(0, __builtin_bit_cast(int, v), CTRL,
                                         0xF, 0xF, true));
}
// 4-stage tree: quad swap1 (0xB1), quad swap2 (0x4E), half mirror, row mirror.
__device__ __forceinline__ float red16_max(float v) {
  v = fmaxf(v, dpp_mov<0xB1>(v));
  v = fmaxf(v, dpp_mov<0x4E>(v));
  v = fmaxf(v, dpp_mov<0x141>(v));
  v = fmaxf(v, dpp_mov<0x140>(v));
  return v;
}
__device__ __forceinline__ float red16_sum(float v) {
  v += dpp_mov<0xB1>(v);
  v += dpp_mov<0x4E>(v);
  v += dpp_mov<0x141>(v);
  v += dpp_mov<0x140>(v);
  return v;
}

// ---- tile loaders ---------------------------------------------------------
// A-matrix tile (16 rows x 32 K), bf16, row-major memory. Element pair
// (2i,2i+1) of the v16bf is dword i; dword index used here (contiguous K order, as in B):
//   pair(i) = i + hf*8   ->  {0..7} (+8 upper half), K contiguous per half
__device__ __forceinline__ v16bf load_a16x32(const __bf16* tile, int rowStride) {
  const int lane = threadIdx.x & 31;
  const int hf = lane >> 4, row = lane & 15;
  const unsigned* p = reinterpret_cast<const unsigned*>(tile + (size_t)row * rowStride);
  v8u u;
#pragma unroll
  for (int i = 0; i < 8; ++i) {
    const int pairIdx = i + hf * 8;
    u[i] = p[pairIdx];
  }
  return __builtin_bit_cast(v16bf, u);
}
// Same A-layout gather from an LDS row base (dword loads).
__device__ __forceinline__ v16bf lds_load_a16x32(const __bf16* rowBase, int hf) {
  const unsigned* p = reinterpret_cast<const unsigned*>(rowBase);
  v8u u;
#pragma unroll
  for (int i = 0; i < 8; ++i) {
    const int pairIdx = i + hf * 8;
    u[i] = p[pairIdx];
  }
  return __builtin_bit_cast(v16bf, u);
}

// B-matrix tile (32 K x 16 cols): lane = column (stride colStride), K contiguous;
// upper lane half takes K=16..31 -> one contiguous 32-byte v16bf load per lane.
__device__ __forceinline__ v16bf load_b32x16(const __bf16* tile, int colStride) {
  const int lane = threadIdx.x & 31;
  const int hf = lane >> 4, n = lane & 15;
  return *reinterpret_cast<const v16bf*>(tile + (size_t)n * colStride + hf * 16);
}

// --------------------------- conversion kernels ----------------------------

__global__ __launch_bounds__(256) void cvt_qk_kernel(const float* __restrict__ Q,
                                                     const float* __restrict__ K,
                                                     __bf16* __restrict__ Qbf,
                                                     __bf16* __restrict__ Kbf) {
  const size_t i = ((size_t)blockIdx.x * 256 + threadIdx.x) * 4;
  const float4 q = *reinterpret_cast<const float4*>(Q + i);
  const float4 k = *reinterpret_cast<const float4*>(K + i);
  uint2 qp, kp;
  qp.x = cvt2bf_u32(q.x * QSCALE, q.y * QSCALE);
  qp.y = cvt2bf_u32(q.z * QSCALE, q.w * QSCALE);
  kp.x = cvt2bf_u32(k.x, k.y);
  kp.y = cvt2bf_u32(k.z, k.w);
  *reinterpret_cast<uint2*>(Qbf + i) = qp;
  *reinterpret_cast<uint2*>(Kbf + i) = kp;
}

__global__ __launch_bounds__(256) void cvt_w_kernel(const float* __restrict__ W,
                                                    __bf16* __restrict__ Wbf) {
  const size_t i = ((size_t)blockIdx.x * 256 + threadIdx.x) * 4;
  const float4 w = *reinterpret_cast<const float4*>(W + i);
  uint2 p;
  p.x = cvt2bf_u32(w.x, w.y);
  p.y = cvt2bf_u32(w.z, w.w);
  *reinterpret_cast<uint2*>(Wbf + i) = p;
}

// V transpose: Vt[(b*512 + h*64 + c)*2048 + n] = bf16(V[(b*2048+n)*512 + h*64 + c])
// 32x32 tiles through LDS so both global read and write stay coalesced.
__global__ __launch_bounds__(256) void cvt_v_kernel(const float* __restrict__ V,
                                                    __bf16* __restrict__ Vt) {
  __shared__ __bf16 tile[32][33];
  const int tid = blockIdx.x;            // 4096 tiles: b(4) x c-tiles(16) x n-tiles(64)
  const int n0 = (tid & 63) * 32;
  const int c0 = ((tid >> 6) & 15) * 32;
  const int b  = tid >> 10;
  const int x  = threadIdx.x & 31;
  const int y0 = threadIdx.x >> 5;       // 0..7
#pragma unroll
  for (int j = 0; j < 4; ++j) {
    const int n = n0 + y0 + j * 8;
    tile[y0 + j * 8][x] = f2bf(V[((size_t)(b * N_ + n)) * D_ + c0 + x]);
  }
  __syncthreads();
#pragma unroll
  for (int j = 0; j < 4; ++j) {
    const int c = c0 + y0 + j * 8;
    Vt[((size_t)(b * D_ + c)) * N_ + n0 + x] = tile[x][y0 + j * 8];
  }
}

// --------------------------- flash attention -------------------------------
// 1 wave == one 16-query tile of one (b,h). Keys in chunks of 64:
// 8 WMMA for S (4 N-tiles, K-depth 64), DPP online softmax, P via per-wave LDS
// slice into A-layout (2 tiles), 8 WMMA for PV. Next chunk prefetched.
__global__ __launch_bounds__(128) void attn_kernel(const __bf16* __restrict__ Qbf,
                                                   const __bf16* __restrict__ Kbf,
                                                   const __bf16* __restrict__ Vt,
                                                   __bf16* __restrict__ Xbf) {
  __shared__ __bf16 ldsP[4][16 * 64];          // 2 KB per wave
  const int lane = threadIdx.x & 31;
  const int wv   = threadIdx.x >> 5;
  const int t    = blockIdx.x * 4 + wv;        // 4096 tiles total
  const int qi   = t & 127;
  const int bh   = t >> 7;                     // b*8 + h
  const int q0   = qi * 16;
  const int hf   = lane >> 4, l16 = lane & 15;
  __bf16* P = ldsP[wv];

  const int b = bh >> 3, h = bh & 7;
  const __bf16* Qbase = Qbf + ((size_t)(b * N_ + q0) * D_ + h * HEAD_);
  const v16bf qa0 = load_a16x32(Qbase, D_);        // head dims 0..31
  const v16bf qa1 = load_a16x32(Qbase + 32, D_);   // head dims 32..63

  v8f o0 = {}, o1 = {}, o2 = {}, o3 = {};
  float mrow[8], lrow[8];
#pragma unroll
  for (int r = 0; r < 8; ++r) { mrow[r] = -__builtin_inff(); lrow[r] = 0.0f; }

  const __bf16* Krow0 = Kbf + ((size_t)b * N_ * D_ + h * HEAD_);
  const __bf16* Vhead = Vt + (size_t)bh * HEAD_ * N_;

  for (int kc = 0; kc < N_; kc += 64) {
    // ---- prefetch next chunk's K rows / Vt rows (2 rows per lane each) ----
    const int kn = (kc + 64 < N_) ? (kc + 64) : kc;
    const __bf16* Kp = Krow0 + (size_t)(kn + 2 * lane) * D_;
    __builtin_prefetch(Kp, 0, 3);
    __builtin_prefetch(Kp + D_, 0, 3);
    const __bf16* Vp = Vhead + (size_t)(2 * lane) * N_ + kn;
    __builtin_prefetch(Vp, 0, 3);
    __builtin_prefetch(Vp + N_, 0, 3);

    // ---- S = Q * K^T : four 16-key N-tiles, K-depth 64 (8 WMMA) ----
    const __bf16* Kb = Krow0 + (size_t)kc * D_;
    v8f s0 = {}, s1 = {}, s2 = {}, s3 = {};
    s0 = wmma_bf16(qa0, load_b32x16(Kb, D_), s0);
    s0 = wmma_bf16(qa1, load_b32x16(Kb + 32, D_), s0);
    s1 = wmma_bf16(qa0, load_b32x16(Kb + 16 * D_, D_), s1);
    s1 = wmma_bf16(qa1, load_b32x16(Kb + 16 * D_ + 32, D_), s1);
    s2 = wmma_bf16(qa0, load_b32x16(Kb + 32 * D_, D_), s2);
    s2 = wmma_bf16(qa1, load_b32x16(Kb + 32 * D_ + 32, D_), s2);
    s3 = wmma_bf16(qa0, load_b32x16(Kb + 48 * D_, D_), s3);
    s3 = wmma_bf16(qa1, load_b32x16(Kb + 48 * D_ + 32, D_), s3);

    // ---- online softmax (rows live in lane halves: row = r + 8*hf) ----
    float arow[8];
#pragma unroll
    for (int r = 0; r < 8; ++r) {
      float v = fmaxf(fmaxf(s0[r], s1[r]), fmaxf(s2[r], s3[r]));
      v = red16_max(v);
      const float mnew  = fmaxf(mrow[r], v);
      const float alpha = exp2f(mrow[r] - mnew);
      mrow[r] = mnew;
      const float p0 = exp2f(s0[r] - mnew);
      const float p1 = exp2f(s1[r] - mnew);
      const float p2 = exp2f(s2[r] - mnew);
      const float p3 = exp2f(s3[r] - mnew);
      const float ls = red16_sum((p0 + p1) + (p2 + p3));
      lrow[r] = lrow[r] * alpha + ls;
      arow[r] = alpha;
      const v2bf q01 = cvt2bf(p0, p1);
      const v2bf q23 = cvt2bf(p2, p3);
      const int row = r + 8 * hf;
      P[row * 64 + l16]      = q01[0];
      P[row * 64 + 16 + l16] = q01[1];
      P[row * 64 + 32 + l16] = q23[0];
      P[row * 64 + 48 + l16] = q23[1];
    }
#pragma unroll
    for (int r = 0; r < 8; ++r) {
      o0[r] *= arow[r]; o1[r] *= arow[r]; o2[r] *= arow[r]; o3[r] *= arow[r];
    }

    // ---- P: C-layout -> two A-layout tiles via per-wave LDS slice ----
    const __bf16* Prow = P + l16 * 64;
    const v16bf pa0 = lds_load_a16x32(Prow, hf);        // chunk keys 0..31
    const v16bf pa1 = lds_load_a16x32(Prow + 32, hf);   // chunk keys 32..63

    // ---- O += P * V : K-depth 64 over keys, 4 col-blocks (8 WMMA) ----
    const __bf16* Vb = Vhead + kc;
    o0 = wmma_bf16(pa0, load_b32x16(Vb, N_), o0);
    o0 = wmma_bf16(pa1, load_b32x16(Vb + 32, N_), o0);
    o1 = wmma_bf16(pa0, load_b32x16(Vb + 16 * N_, N_), o1);
    o1 = wmma_bf16(pa1, load_b32x16(Vb + 16 * N_ + 32, N_), o1);
    o2 = wmma_bf16(pa0, load_b32x16(Vb + 32 * N_, N_), o2);
    o2 = wmma_bf16(pa1, load_b32x16(Vb + 32 * N_ + 32, N_), o2);
    o3 = wmma_bf16(pa0, load_b32x16(Vb + 48 * N_, N_), o3);
    o3 = wmma_bf16(pa1, load_b32x16(Vb + 48 * N_ + 32, N_), o3);
  }

  // ---- normalize and write bf16 attention output (B,N,D layout) ----
#pragma unroll
  for (int r = 0; r < 8; ++r) {
    const float inv = 1.0f / lrow[r];
    __bf16* X = Xbf + ((size_t)(b * N_ + q0 + 8 * hf + r) * D_ + h * HEAD_ + l16);
    const v2bf x01 = cvt2bf(o0[r] * inv, o1[r] * inv);
    const v2bf x23 = cvt2bf(o2[r] * inv, o3[r] * inv);
    X[0]  = x01[0];
    X[16] = x01[1];
    X[32] = x23[0];
    X[48] = x23[1];
  }
}

// --------------------------- output projection -----------------------------
// out = X @ W^T + b : M=8192, N=512, K=512. Wave tile 16x64, bf16 WMMA.
// B-column j of (W^T) is row j of W -> K contiguous in the original W layout.
__global__ __launch_bounds__(128) void proj_kernel(const __bf16* __restrict__ Xbf,
                                                   const __bf16* __restrict__ Wbf,
                                                   const float* __restrict__ bias,
                                                   float* __restrict__ out) {
  const int lane = threadIdx.x & 31;
  const int wv = threadIdx.x >> 5;
  const int t  = blockIdx.x * 4 + wv;          // 4096 tiles: 512 row x 8 col
  const int ct = t & 7, rt = t >> 3;
  const int m0 = rt * 16, n0 = ct * 64;
  const int hf = lane >> 4, l16 = lane & 15;

  v8f c0 = {}, c1 = {}, c2 = {}, c3 = {};
  for (int k0 = 0; k0 < D_; k0 += 32) {
    const v16bf a = load_a16x32(Xbf + (size_t)m0 * D_ + k0, D_);
    c0 = wmma_bf16(a, load_b32x16(Wbf + (size_t)(n0 + 0)  * D_ + k0, D_), c0);
    c1 = wmma_bf16(a, load_b32x16(Wbf + (size_t)(n0 + 16) * D_ + k0, D_), c1);
    c2 = wmma_bf16(a, load_b32x16(Wbf + (size_t)(n0 + 32) * D_ + k0, D_), c2);
    c3 = wmma_bf16(a, load_b32x16(Wbf + (size_t)(n0 + 48) * D_ + k0, D_), c3);
  }
  const float b0 = bias[n0 + l16];
  const float b1 = bias[n0 + 16 + l16];
  const float b2 = bias[n0 + 32 + l16];
  const float b3 = bias[n0 + 48 + l16];
#pragma unroll
  for (int r = 0; r < 8; ++r) {
    float* o = out + (size_t)(m0 + 8 * hf + r) * D_ + n0 + l16;
    o[0]  = c0[r] + b0;
    o[16] = c1[r] + b1;
    o[32] = c2[r] + b2;
    o[48] = c3[r] + b3;
  }
}

// ------------------------------- launcher ----------------------------------

extern "C" void kernel_launch(void* const* d_in, const int* in_sizes, int n_in,
                              void* d_out, int out_size, void* d_ws, size_t ws_size,
                              hipStream_t stream) {
  (void)in_sizes; (void)n_in; (void)out_size; (void)ws_size;
  const float* keys    = (const float*)d_in[0];
  const float* queries = (const float*)d_in[1];
  const float* values  = (const float*)d_in[2];
  const float* Wc      = (const float*)d_in[3];
  const float* bc      = (const float*)d_in[4];
  float* out = (float*)d_out;

  // workspace layout (bf16 staging): Q | K | Vt | X | W  = 4*8MB + 0.5MB
  char* ws = (char*)d_ws;
  const size_t QKV = (size_t)B_ * N_ * D_ * sizeof(__bf16);   // 8 MB
  __bf16* Qbf = (__bf16*)(ws);
  __bf16* Kbf = (__bf16*)(ws + QKV);
  __bf16* Vt  = (__bf16*)(ws + 2 * QKV);
  __bf16* Xbf = (__bf16*)(ws + 3 * QKV);
  __bf16* Wbf = (__bf16*)(ws + 4 * QKV);

  const int elemsQK = B_ * N_ * D_;                       // 4,194,304
  cvt_qk_kernel<<<dim3(elemsQK / (4 * 256)), dim3(256), 0, stream>>>(queries, keys, Qbf, Kbf);
  cvt_v_kernel<<<dim3(4096), dim3(256), 0, stream>>>(values, Vt);
  cvt_w_kernel<<<dim3((D_ * D_) / (4 * 256)), dim3(256), 0, stream>>>(Wc, Wbf);

  // 4096 query-tiles (B*H * N/16), 4 waves per block
  attn_kernel<<<dim3(1024), dim3(128), 0, stream>>>(Qbf, Kbf, Vt, Xbf);

  // 4096 GEMM tiles (8192/16 rows x 512/64 cols), 4 waves per block
  proj_kernel<<<dim3(1024), dim3(128), 0, stream>>>(Xbf, Wbf, bc, out);
}
